// RadialBasisMapping_6562710028922
// MI455X (gfx1250) — compile-verified, matrix operand order checked by reading, not run on hardware
//
#include <hip/hip_runtime.h>
#include <hip/hip_bf16.h>

// ---------------------------------------------------------------------------
// RBF mapping: out[b,k] = exp(-beta[k] * (||x_b||^2 + ||c_k||^2 - 2 x_b.c_k))
// B=2048, K=1024, D=49152.  GEMM via v_wmma_f32_16x16x32_bf16, norms fused
// into the fp32->bf16 conversion pass, exp epilogue fused into the store.
// ---------------------------------------------------------------------------

typedef __attribute__((ext_vector_type(16))) __bf16 v16bf;
typedef __attribute__((ext_vector_type(8)))  float  v8f;

#define B_SZ 2048
#define K_SZ 1024
#define D_SZ 49152
#define BM   128     // x rows per block
#define BN   64      // center rows per block
#define BK   32      // K-slice per stage (one bf16 WMMA K-step)
#define NTHREADS 256 // 8 waves

union FragU {
    uint4 q[2];   // two 16B chunks per ISA bf16 A/B fragment layout
    v16bf v;
};

// round-to-nearest-even fp32 -> bf16, packed pair into one dword
__device__ __forceinline__ unsigned int pack2_bf16(float a, float b) {
    unsigned int ua = __float_as_uint(a);
    unsigned int ub = __float_as_uint(b);
    unsigned int lo = (ua + 0x7FFFu + ((ua >> 16) & 1u)) >> 16;
    unsigned int hi = (ub + 0x7FFFu + ((ub >> 16) & 1u)) & 0xFFFF0000u;
    return lo | hi;
}

__global__ __launch_bounds__(NTHREADS)
void rbf_wmma_kernel(const float* __restrict__ x,
                     const float* __restrict__ centers,
                     const float* __restrict__ beta,
                     float* __restrict__ out)
{
    __shared__ __align__(16) unsigned short sA[BM * BK]; // 8 KB bf16 x tile
    __shared__ __align__(16) unsigned short sB[BN * BK]; // 4 KB bf16 c tile
    __shared__ float sXpart[NTHREADS];
    __shared__ float sCpart[NTHREADS];
    __shared__ float sXsq[BM];
    __shared__ float sCsq[BN];

    const int tid  = threadIdx.x;
    const int lane = tid & 31;
    const int wave = tid >> 5;   // 0..7
    const int wm   = wave >> 1;  // 0..3  (32-row band)
    const int wn   = wave & 1;   // 0..1  (32-col band)

    const int m0 = blockIdx.y * BM;  // global x-row base
    const int n0 = blockIdx.x * BN;  // global center-row base

    // --- cooperative load mapping (fp32) ---------------------------------
    // x tile: 128 rows x 32 cols -> 16 floats / thread (2 threads per row)
    const int axRow = tid >> 1;
    const int axCol = (tid & 1) * 16;
    // c tile: 64 rows x 32 cols  -> 8 floats / thread (4 threads per row)
    const int acRow = tid >> 2;
    const int acCol = (tid & 3) * 8;

    const float* xBase = x       + (size_t)(m0 + axRow) * D_SZ + axCol;
    const float* cBase = centers + (size_t)(n0 + acRow) * D_SZ + acCol;

    float xsq = 0.0f;
    float csq = 0.0f;

    v8f acc[2][2];
    #pragma unroll
    for (int i = 0; i < 2; ++i)
        #pragma unroll
        for (int j = 0; j < 2; ++j)
            acc[i][j] = (v8f){0.f, 0.f, 0.f, 0.f, 0.f, 0.f, 0.f, 0.f};

    const int fr = lane & 15;          // M (A) / N (B) within 16x16 tile
    const int kh = (lane >> 4) * 8;    // K-group select within fragment

    for (int d0 = 0; d0 < D_SZ; d0 += BK) {
        // ---- global fp32 loads (latency overlaps the barrier) -----------
        float xr[16];
        #pragma unroll
        for (int i = 0; i < 4; ++i)
            *(float4*)&xr[4 * i] = *(const float4*)(xBase + d0 + 4 * i);
        float cr[8];
        #pragma unroll
        for (int i = 0; i < 2; ++i)
            *(float4*)&cr[4 * i] = *(const float4*)(cBase + d0 + 4 * i);

        // prefetch next K-slice (global_prefetch_b8; speculative, OOB-safe)
        __builtin_prefetch(xBase + d0 + BK, 0, 1);
        __builtin_prefetch(cBase + d0 + BK, 0, 1);

        __syncthreads();  // previous stage's LDS reads complete

        // ---- fused norm accumulation + bf16 convert + LDS stage ---------
        unsigned int wx[8];
        #pragma unroll
        for (int j = 0; j < 8; ++j) {
            xsq = fmaf(xr[2 * j], xr[2 * j], fmaf(xr[2 * j + 1], xr[2 * j + 1], xsq));
            wx[j] = pack2_bf16(xr[2 * j], xr[2 * j + 1]);
        }
        *(uint4*)&sA[axRow * BK + axCol]     = *(uint4*)&wx[0];
        *(uint4*)&sA[axRow * BK + axCol + 8] = *(uint4*)&wx[4];

        unsigned int wc[4];
        #pragma unroll
        for (int j = 0; j < 4; ++j) {
            csq = fmaf(cr[2 * j], cr[2 * j], fmaf(cr[2 * j + 1], cr[2 * j + 1], csq));
            wc[j] = pack2_bf16(cr[2 * j], cr[2 * j + 1]);
        }
        *(uint4*)&sB[acRow * BK + acCol] = *(uint4*)&wc[0];

        __syncthreads();  // tiles visible to all waves

        // ---- fragment loads (2x ds_read_b128 each, per ISA layout) ------
        FragU af[2], bf[2];
        #pragma unroll
        for (int mt = 0; mt < 2; ++mt) {
            const int row = wm * 32 + mt * 16 + fr;
            af[mt].q[0] = *(const uint4*)&sA[row * BK + kh];
            af[mt].q[1] = *(const uint4*)&sA[row * BK + 16 + kh];
        }
        #pragma unroll
        for (int nt = 0; nt < 2; ++nt) {
            const int col = wn * 32 + nt * 16 + fr;
            bf[nt].q[0] = *(const uint4*)&sB[col * BK + kh];
            bf[nt].q[1] = *(const uint4*)&sB[col * BK + 16 + kh];
        }

        // ---- matrix math: 4x v_wmma_f32_16x16x32_bf16 per wave ----------
        #pragma unroll
        for (int mt = 0; mt < 2; ++mt)
            #pragma unroll
            for (int nt = 0; nt < 2; ++nt)
                acc[mt][nt] = __builtin_amdgcn_wmma_f32_16x16x32_bf16(
                    /*neg_a=*/false, af[mt].v,
                    /*neg_b=*/false, bf[nt].v,
                    /*c_mod=*/(short)0, acc[mt][nt],
                    /*reuse_a=*/false, /*reuse_b=*/false);
    }

    // ---- reduce partial squared norms to per-row / per-col values -------
    sXpart[tid] = xsq;
    sCpart[tid] = csq;
    __syncthreads();
    if (tid < BM) sXsq[tid] = sXpart[2 * tid] + sXpart[2 * tid + 1];
    if (tid < BN) sCsq[tid] = sCpart[4 * tid]     + sCpart[4 * tid + 1] +
                              sCpart[4 * tid + 2] + sCpart[4 * tid + 3];
    __syncthreads();

    // ---- epilogue: dist -> exp(-beta*dist), store -----------------------
    // C/D layout: VGPR r, lanes 0-15 -> M = r, lanes 16-31 -> M = 8 + r; N = lane&15
    #pragma unroll
    for (int mt = 0; mt < 2; ++mt) {
        #pragma unroll
        for (int nt = 0; nt < 2; ++nt) {
            const int rbase = wm * 32 + mt * 16 + (lane >> 4) * 8;
            const int col_l = wn * 32 + nt * 16 + (lane & 15);
            const int col_g = n0 + col_l;
            const float bet = beta[col_g];
            const float cs  = sCsq[col_l];
            #pragma unroll
            for (int r = 0; r < 8; ++r) {
                const int   row_l = rbase + r;
                const float dist  = sXsq[row_l] + cs - 2.0f * acc[mt][nt][r];
                out[(size_t)(m0 + row_l) * K_SZ + col_g] = __expf(-bet * dist);
            }
        }
    }
}

extern "C" void kernel_launch(void* const* d_in, const int* in_sizes, int n_in,
                              void* d_out, int out_size, void* d_ws, size_t ws_size,
                              hipStream_t stream) {
    const float* x       = (const float*)d_in[0];
    const float* centers = (const float*)d_in[1];
    const float* beta    = (const float*)d_in[2];
    float* out = (float*)d_out;

    dim3 grid(K_SZ / BN, B_SZ / BM);   // 16 x 16 = 256 blocks
    dim3 block(NTHREADS);
    hipLaunchKernelGGL(rbf_wmma_kernel, grid, block, 0, stream,
                       x, centers, beta, out);

    (void)in_sizes; (void)n_in; (void)out_size; (void)d_ws; (void)ws_size;
}
